// GCN_20864951124337
// MI455X (gfx1250) — compile-verified, matrix operand order checked by reading, not run on hardware
//
#include <hip/hip_runtime.h>
#include <hip/hip_bf16.h>

#define Nn 50000
#define Ee 800000
#define Hh 128
#define Ll 4
#define Cc 128
#define EFf 7
#define BN_EPS 1e-5f

typedef __attribute__((ext_vector_type(16))) __bf16 v16bf;
typedef __attribute__((ext_vector_type(8)))  __bf16 v8bf;
typedef __attribute__((ext_vector_type(8)))  float  v8f;

// ---------------- utility ----------------
__global__ void zero_kernel(float* __restrict__ p, int n) {
    int i = blockIdx.x * blockDim.x + threadIdx.x;
    if (i < n) p[i] = 0.0f;
}

// h_bf[i,c] = (bf16) node_emb[nfeat[i], c]
__global__ void init_h_kernel(const int* __restrict__ nfeat,
                              const float* __restrict__ node_emb,
                              __bf16* __restrict__ hbf) {
    int idx = blockIdx.x * blockDim.x + threadIdx.x;
    if (idx >= Nn * Hh) return;
    int i = idx >> 7;          // /128
    int c = idx & 127;
    hbf[idx] = (__bf16)node_emb[(size_t)nfeat[i] * Hh + c];
}

// ---------------- pre-pack W (all 4 layers) into WMMA B-fragment layout, bf16 ----------------
// Fragment storage: Wf[((l*8 + tc)*4 + ks)*32 + lane][16]  (16 contiguous bf16 per lane)
__global__ void pack_W_kernel(const float* __restrict__ W,   // L x 128 x 128 (k, n)
                              __bf16* __restrict__ Wf) {
    int t = blockIdx.x * blockDim.x + threadIdx.x;
    if (t >= Ll * Hh * Hh) return;
    int i    = t & 15;          // element within lane fragment
    int lane = (t >> 4) & 31;
    int ks   = (t >> 9) & 3;    // K-step (32 per step)
    int tc   = (t >> 11) & 7;   // column tile
    int l    = t >> 14;         // layer
    int n     = tc * 16 + (lane & 15);
    int halfK = (lane >> 4) * 8;
    int k     = ks * 32 + ((i < 8) ? (halfK + i) : (16 + halfK + (i - 8)));
    Wf[t] = (__bf16)W[(size_t)l * Hh * Hh + (size_t)k * Hh + n];
}

// ---------------- WMMA GEMM: hl = h @ W + b  (N x 128) @ (128 x 128) ----------------
// One wave computes one 16x16 tile of hl with 4x v_wmma_f32_16x16x32_bf16.
// A from bf16 h (two b128 loads / K-step), B from pre-packed fragments (two b128 loads / K-step).
__global__ void gemm_hl_kernel(const __bf16* __restrict__ hbf,  // N x 128 bf16 row-major
                               const __bf16* __restrict__ Wf,   // fragment-packed, this layer
                               const float* __restrict__ b,     // 128
                               float* __restrict__ hl) {
    const int waveG = blockIdx.x * (blockDim.x >> 5) + (threadIdx.x >> 5);
    const int lane  = threadIdx.x & 31;
    const int tileC = waveG & 7;                  // 8 column tiles
    const int tileR = waveG >> 3;
    if (tileR >= Nn / 16) return;
    const int row0 = tileR * 16;
    const int col0 = tileC * 16;
    const int m     = lane & 15;
    const int halfK = (lane >> 4) * 8;            // 0 or 8

    v8f acc = {};
    const __bf16* ap = hbf + (size_t)(row0 + m) * Hh;

    #pragma unroll
    for (int ks = 0; ks < 4; ++ks) {
        const int k0 = ks * 32;
        // A 16x32 bf16: lanes 0-15 K={0..7,16..23}, lanes 16-31 K={8..15,24..31}
        v8bf alo = *(const v8bf*)(ap + k0 + halfK);
        v8bf ahi = *(const v8bf*)(ap + k0 + 16 + halfK);
        v16bf a = __builtin_shufflevector(alo, ahi,
                    0,1,2,3,4,5,6,7,8,9,10,11,12,13,14,15);
        const __bf16* bp = Wf + ((size_t)(tileC * 4 + ks) * 32 + lane) * 16;
        v8bf blo = *(const v8bf*)(bp);
        v8bf bhi = *(const v8bf*)(bp + 8);
        v16bf bm = __builtin_shufflevector(blo, bhi,
                    0,1,2,3,4,5,6,7,8,9,10,11,12,13,14,15);
        acc = __builtin_amdgcn_wmma_f32_16x16x32_bf16(
            /*neg_a=*/false, a, /*neg_b=*/false, bm,
            /*c_mod=*/(short)0, acc, /*reuse_a=*/false, /*reuse_b=*/false);
    }

    // C/D layout: VGPR r holds row (r + 8*(lane>=16)), col = lane&15
    const int cc     = col0 + (lane & 15);
    const float bias = b[cc];
    const int rbase  = row0 + ((lane >> 4) << 3);
    #pragma unroll
    for (int r = 0; r < 8; ++r)
        hl[(size_t)(rbase + r) * Hh + cc] = acc[r] + bias;
}

// ---------------- edge phase: agg[dst] += enorm * relu(hl[src] + efeat@eW + eb) ----------------
// One wave (32 lanes) per edge, 4 channels/lane (float4). Working set is L2-resident.
__global__ void edge_kernel(const int* __restrict__ src,
                            const int* __restrict__ dst,
                            const float* __restrict__ efeat,   // E x 7
                            const float* __restrict__ degs,    // N
                            const float* __restrict__ hl,      // N x 128
                            const float* __restrict__ eW,      // 7 x 128
                            const float* __restrict__ eb,      // 128
                            float* __restrict__ agg) {
    __shared__ float s_eW[EFf * Hh];
    __shared__ float s_eb[Hh];
    for (int i = threadIdx.x; i < EFf * Hh; i += blockDim.x) s_eW[i] = eW[i];
    for (int i = threadIdx.x; i < Hh; i += blockDim.x)       s_eb[i] = eb[i];
    __syncthreads();

    const int wave = threadIdx.x >> 5;
    const int lane = threadIdx.x & 31;
    const int e = blockIdx.x * 8 + wave;
    if (e >= Ee) return;

    const int s = src[e];
    const int d = dst[e];
    const float enorm = __frsqrt_rn(degs[s]) * __frsqrt_rn(degs[d]);

    const int c0 = lane << 2;
    // prefetch the gather row of an edge ~64 ahead (global_prefetch_b8)
    const int epf = e + 64;
    if (epf < Ee) {
        int spf = src[epf];
        __builtin_prefetch(hl + (size_t)spf * Hh + c0, 0, 1);
    }

    float ef[EFf];
    #pragma unroll
    for (int k = 0; k < EFf; ++k) ef[k] = efeat[(size_t)e * EFf + k];

    const float4 hv = *(const float4*)(hl + (size_t)s * Hh + c0);
    const float hvv[4] = {hv.x, hv.y, hv.z, hv.w};
    float* aggp = agg + (size_t)d * Hh + c0;

    #pragma unroll
    for (int j = 0; j < 4; ++j) {
        const int c = c0 + j;
        float ev = s_eb[c];
        #pragma unroll
        for (int k = 0; k < EFf; ++k) ev += ef[k] * s_eW[k * Hh + c];
        float msg = hvv[j] + ev;
        msg = msg > 0.0f ? msg : 0.0f;
        unsafeAtomicAdd(aggp + j, enorm * msg);   // global_atomic_add_f32 in L2
    }
}

// ---------------- node update: hn = agg + relu(hl + root)/deg ; accumulate BN stats ----------------
#define NODES_PER_BLOCK 128
__global__ void node_update_kernel(const float* __restrict__ hl,
                                   float* __restrict__ agg,       // in: agg, out: hn (in place)
                                   const float* __restrict__ degs,
                                   const float* __restrict__ root,
                                   float* __restrict__ colsum,
                                   float* __restrict__ colsumsq) {
    const int c = threadIdx.x;   // 128 threads = 4 waves, one channel each
    const float rt = root[c];
    float s = 0.0f, s2 = 0.0f;
    const int i0 = blockIdx.x * NODES_PER_BLOCK;
    const int i1 = min(i0 + NODES_PER_BLOCK, Nn);
    for (int i = i0; i < i1; ++i) {
        const float inv = 1.0f / degs[i];
        float r = hl[(size_t)i * Hh + c] + rt;
        r = r > 0.0f ? r : 0.0f;
        const float hn = agg[(size_t)i * Hh + c] + r * inv;
        agg[(size_t)i * Hh + c] = hn;
        s  += hn;
        s2 += hn * hn;
    }
    unsafeAtomicAdd(&colsum[c],   s);
    unsafeAtomicAdd(&colsumsq[c], s2);
}

// ---------------- BN apply (+optional relu) ----------------
// Layers 0..2: write bf16 h for the next WMMA GEMM.
// Last layer: skip the store, accumulate graph-pool column sums directly.
__global__ void bn_apply_kernel(const float* __restrict__ hn,
                                __bf16* __restrict__ hbf,
                                const float* __restrict__ colsum,
                                const float* __restrict__ colsumsq,
                                const float* __restrict__ gamma,
                                const float* __restrict__ beta,
                                int do_relu,
                                float* __restrict__ pool) {    // nullptr except last layer
    const int c = threadIdx.x;
    const float invN = 1.0f / (float)Nn;
    const float mu   = colsum[c] * invN;
    const float var  = colsumsq[c] * invN - mu * mu;
    const float rstd = __frsqrt_rn(var + BN_EPS);
    const float g = gamma[c], bt = beta[c];
    const bool doPool = (pool != nullptr);
    float s = 0.0f;
    const int i0 = blockIdx.x * NODES_PER_BLOCK;
    const int i1 = min(i0 + NODES_PER_BLOCK, Nn);
    for (int i = i0; i < i1; ++i) {
        float v = g * (hn[(size_t)i * Hh + c] - mu) * rstd + bt;
        if (do_relu) v = v > 0.0f ? v : 0.0f;
        if (doPool) s += v;
        else        hbf[(size_t)i * Hh + c] = (__bf16)v;
    }
    if (doPool) unsafeAtomicAdd(&pool[c], s);
}

// ---------------- classifier: out = (pool/N) @ Wp + bp ----------------
__global__ void classifier_kernel(const float* __restrict__ pool,
                                  const float* __restrict__ Wp,   // 128 x 128
                                  const float* __restrict__ bp,
                                  float* __restrict__ out) {
    const int c = threadIdx.x;   // 128 threads
    const float invN = 1.0f / (float)Nn;
    float acc = bp[c];
    for (int k = 0; k < Hh; ++k)
        acc += (pool[k] * invN) * Wp[(size_t)k * Cc + c];
    out[c] = acc;
}

// ---------------- host orchestration ----------------
extern "C" void kernel_launch(void* const* d_in, const int* in_sizes, int n_in,
                              void* d_out, int out_size, void* d_ws, size_t ws_size,
                              hipStream_t stream) {
    const int*   nfeat    = (const int*)  d_in[0];
    const int*   src      = (const int*)  d_in[1];
    const int*   dst      = (const int*)  d_in[2];
    const float* efeat    = (const float*)d_in[3];
    const float* degs     = (const float*)d_in[4];
    const float* node_emb = (const float*)d_in[5];
    const float* W        = (const float*)d_in[6];   // L x 128 x 128
    const float* b        = (const float*)d_in[7];   // L x 128
    const float* root     = (const float*)d_in[8];   // L x 128
    const float* eW       = (const float*)d_in[9];   // L x 7 x 128
    const float* eb       = (const float*)d_in[10];  // L x 128
    const float* gamma    = (const float*)d_in[11];  // L x 128
    const float* beta     = (const float*)d_in[12];  // L x 128
    const float* Wp       = (const float*)d_in[13];  // 128 x 128
    const float* bp       = (const float*)d_in[14];  // 128
    float* out = (float*)d_out;

    const size_t NH = (size_t)Nn * Hh;               // 6.4M
    float*  hl       = (float*)d_ws;
    float*  agg      = hl + NH;
    float*  colsum   = agg + NH;                     // agg..stats contiguous (one zero pass)
    float*  colsumsq = colsum + Hh;
    float*  pool     = colsumsq + Hh;
    __bf16* Wf       = (__bf16*)(pool + Hh);         // 4*128*128 bf16 (16B-aligned)
    __bf16* hbf      = Wf + (size_t)Ll * Hh * Hh;    // N*128 bf16

    // one-time: pack weights into WMMA B fragments; h0 = node_emb[nfeat]
    {
        int n = Ll * Hh * Hh;                        // 65536
        pack_W_kernel<<<(n + 255) / 256, 256, 0, stream>>>(W, Wf);
        int nh = (int)NH;
        init_h_kernel<<<(nh + 255) / 256, 256, 0, stream>>>(nfeat, node_emb, hbf);
    }

    const int nodeBlocks = (Nn + NODES_PER_BLOCK - 1) / NODES_PER_BLOCK;
    for (int l = 0; l < Ll; ++l) {
        const bool last = (l == Ll - 1);
        // zero agg + colsum + colsumsq + pool in one contiguous pass
        {
            int n = (int)(NH + 3 * Hh);
            zero_kernel<<<(n + 255) / 256, 256, 0, stream>>>(agg, n);
        }
        // hl = h @ W[l] + b[l]   (WMMA bf16, f32 accumulate)
        {
            int waves  = (Nn / 16) * 8;              // 25000
            int blocks = waves / 8;                  // 8 waves / 256-thread block
            gemm_hl_kernel<<<blocks, 256, 0, stream>>>(
                hbf, Wf + (size_t)l * Hh * Hh, b + (size_t)l * Hh, hl);
        }
        // edge gather + encode + scatter-add
        edge_kernel<<<Ee / 8, 256, 0, stream>>>(
            src, dst, efeat, degs, hl,
            eW + (size_t)l * EFf * Hh, eb + (size_t)l * Hh, agg);
        // hn (in place over agg) + BN stats
        node_update_kernel<<<nodeBlocks, Hh, 0, stream>>>(
            hl, agg, degs, root + (size_t)l * Hh, colsum, colsumsq);
        // BN apply: layers 0..2 write bf16 h; last layer pools directly
        bn_apply_kernel<<<nodeBlocks, Hh, 0, stream>>>(
            agg, hbf, colsum, colsumsq,
            gamma + (size_t)l * Hh, beta + (size_t)l * Hh,
            last ? 0 : 1, last ? pool : nullptr);
    }

    classifier_kernel<<<1, Cc, 0, stream>>>(pool, Wp, bp, out);
    (void)in_sizes; (void)n_in; (void)out_size; (void)ws_size;
}
